// EMP_64235530879096
// MI455X (gfx1250) — matrix operand order repaired and checked by reading, not compiled, not run on hardware
//
#include <hip/hip_runtime.h>

#define NN 100000
#define DD 128
#define HH 64
#define CC 40
#define RR 3
#define EE 1000000

typedef __attribute__((ext_vector_type(16))) _Float16 v16h;
typedef __attribute__((ext_vector_type(8)))  _Float16 v8h;
typedef __attribute__((ext_vector_type(8)))  float    v8f;

// ---------------- utility kernels ----------------

__global__ void k_zero(float* __restrict__ p, int n) {
    int i = blockIdx.x * blockDim.x + threadIdx.x;
    if (i < n) p[i] = 0.0f;
}

__global__ void k_deg(const int* __restrict__ dst, float* __restrict__ deg, int total) {
    int i = blockIdx.x * blockDim.x + threadIdx.x;
    if (i >= total) return;
    int r = i / EE;
    atomicAdd(&deg[r * NN + dst[i]], 1.0f);
}

__global__ void k_invdeg(float* __restrict__ deg, int n) {
    int i = blockIdx.x * blockDim.x + threadIdx.x;
    if (i >= n) return;
    float d = deg[i];
    deg[i] = (d > 0.0f) ? (1.0f / d) : 0.0f;
}

__global__ void k_f32_to_f16(const float* __restrict__ s, _Float16* __restrict__ d, int n) {
    int i = blockIdx.x * blockDim.x + threadIdx.x;
    if (i < n) d[i] = (_Float16)s[i];
}

__global__ void k_relu_mean_f16(const float* __restrict__ agg, _Float16* __restrict__ h16, int n) {
    int i = blockIdx.x * blockDim.x + threadIdx.x;
    if (i >= n) return;
    float v = agg[i] * (1.0f / (float)RR);
    h16[i] = (_Float16)(v > 0.0f ? v : 0.0f);
}

// Transpose + convert weights: W (f32, [R][K][Cout]) -> Wt (f16, [R][CoutPad][K]), zero-padded cols.
__global__ void k_prep_w(const float* __restrict__ W, _Float16* __restrict__ Wt,
                         int K, int Cout, int CoutPad, int total /* R*CoutPad*K */) {
    int i = blockIdx.x * blockDim.x + threadIdx.x;
    if (i >= total) return;
    int r   = i / (CoutPad * K);
    int rem = i - r * (CoutPad * K);
    int c   = rem / K;
    int k   = rem - c * K;
    float v = (c < Cout) ? W[(size_t)r * K * Cout + (size_t)k * Cout + c] : 0.0f;
    Wt[i] = (_Float16)v;
}

// ---------------- WMMA GEMM: out[M x Cout] = A[M x K](f16) * Wt^T + bias ----------------
// Wt is f16 [CoutPad][K] (pre-transposed). One wave32 computes one 16x16 tile; K fully unrolled.
// A fragment (16-bit A 16x32): row = lane%16; halves 0..7 = K(kk + 8*hi + 0..7),
//   halves 8..15 = K(kk + 8*hi + 16..23), hi = lane>=16.
// B fragment (16-bit B 32x16): col = lane%16; same K order as A: halves 0..7 = Wt[col][kk+8*hi+0..7],
//   halves 8..15 = Wt[col][kk+8*hi+16..23], two 16B loads.
// D (f32 16x16): col = lane%16, row = g + 8*hi for acc element g.
template <int K>
__global__ void k_gemm_wmma(const _Float16* __restrict__ A,
                            const _Float16* __restrict__ Wt,
                            const float* __restrict__ bias,
                            float* __restrict__ out,
                            int Cout, int cTiles, int totalTiles) {
    int gid  = blockIdx.x * blockDim.x + threadIdx.x;
    int wave = gid >> 5;
    int lane = threadIdx.x & 31;
    if (wave >= totalTiles) return;   // wave-uniform exit: EXEC all-ones for live waves

    int mt   = wave / cTiles;
    int ct   = wave - mt * cTiles;
    int ln16 = lane & 15;
    int hi   = lane >> 4;             // 0 or 1
    int arow = mt * 16 + ln16;
    int col  = ct * 16 + ln16;        // always < CoutPad

    v8f acc;
    float bv = (col < Cout) ? bias[col] : 0.0f;
#pragma unroll
    for (int g = 0; g < 8; ++g) acc[g] = bv;

    const _Float16* ap = A  + (size_t)arow * K + hi * 8;
    const _Float16* bp = Wt + (size_t)col  * K + hi * 8;

#pragma unroll
    for (int kk = 0; kk < K; kk += 32) {
        v8h a0 = *(const v8h*)(ap + kk);
        v8h a1 = *(const v8h*)(ap + kk + 16);
        v8h b0 = *(const v8h*)(bp + kk);
        v8h b1 = *(const v8h*)(bp + kk + 16);
        v16h a, b;
#pragma unroll
        for (int j = 0; j < 8; ++j) {
            a[j] = a0[j]; a[8 + j] = a1[j];
            b[j] = b0[j]; b[8 + j] = b1[j];
        }
        acc = __builtin_amdgcn_wmma_f32_16x16x32_f16(
            /*neg_a=*/false, a, /*neg_b=*/false, b,
            /*c_mod=*/(short)0, acc, /*reuse_a=*/false, /*reuse_b=*/false);
    }

    if (col < Cout) {
        int rbase = mt * 16 + hi * 8;
#pragma unroll
        for (int g = 0; g < 8; ++g) out[(size_t)(rbase + g) * Cout + col] = acc[g];
    }
}

// ---------------- degree-normalized scatter ----------------
// out[dst[e]*F + f] += h[src[e]*F + f] * invdeg[dst[e]] * scale
// (edge, feature) lane mapping: contiguous per-edge rows -> coalesced gather + L2 atomics.
template <int F>
__global__ void k_spmm(const float* __restrict__ h,
                       const int* __restrict__ src,
                       const int* __restrict__ dst,
                       const float* __restrict__ invdeg,
                       float* __restrict__ out,
                       float scale, int total) {
    int gid = blockIdx.x * blockDim.x + threadIdx.x;
    if (gid >= total) return;
    int e = gid / F;           // F compile-time: shifts (64) or magic-mul (40)
    int f = gid - e * F;
    int s = src[e];
    int d = dst[e];
    float w = invdeg[d];
    if (w != 0.0f) {
        atomicAdd(&out[(size_t)d * F + f], h[(size_t)s * F + f] * (w * scale));
    }
}

// ---------------- host launch ----------------

extern "C" void kernel_launch(void* const* d_in, const int* in_sizes, int n_in,
                              void* d_out, int out_size, void* d_ws, size_t ws_size,
                              hipStream_t stream) {
    (void)in_sizes; (void)n_in; (void)out_size; (void)ws_size;

    const float* x   = (const float*)d_in[0];   // [N, D]
    const float* W1  = (const float*)d_in[1];   // [R, D, H]
    const float* b1  = (const float*)d_in[2];   // [R, H]
    const float* W2  = (const float*)d_in[3];   // [R, H, C]
    const float* b2  = (const float*)d_in[4];   // [R, C]
    const int* esrc  = (const int*)d_in[5];     // [R, E]
    const int* edst  = (const int*)d_in[6];     // [R, E]
    float* outp      = (float*)d_out;           // [N, C]

    const int C2PAD = 48;                       // layer-2 Cout padded to 16

    // workspace layout (all 16B aligned)
    char* w = (char*)d_ws;
    float*    invdeg = (float*)w;     w += (size_t)RR * NN * sizeof(float);       // 1.2 MB
    float*    hbuf   = (float*)w;     w += (size_t)NN * HH * sizeof(float);       // 25.6 MB (reused as N x C in layer 2)
    float*    agg    = (float*)w;     w += (size_t)NN * HH * sizeof(float);       // 25.6 MB
    _Float16* xh     = (_Float16*)w;  w += (size_t)NN * DD * sizeof(_Float16);    // 25.6 MB
    _Float16* h16    = (_Float16*)w;  w += (size_t)NN * HH * sizeof(_Float16);    // 12.8 MB
    _Float16* wt1    = (_Float16*)w;  w += (size_t)RR * HH * DD * sizeof(_Float16);   // 48 KB  [R][64][128]
    _Float16* wt2    = (_Float16*)w;  /* [R][48][64], 18 KB */

    const int B = 256;
    const int mTiles = NN / 16;  // 6250

    // zero accumulators (harness poisons d_out/d_ws)
    k_zero<<<(RR * NN + B - 1) / B, B, 0, stream>>>(invdeg, RR * NN);
    k_zero<<<(NN * HH + B - 1) / B, B, 0, stream>>>(agg, NN * HH);
    k_zero<<<(NN * CC + B - 1) / B, B, 0, stream>>>(outp, NN * CC);

    // degrees (dst arrays identical across layers), then invert
    k_deg<<<(RR * EE + B - 1) / B, B, 0, stream>>>(edst, invdeg, RR * EE);
    k_invdeg<<<(RR * NN + B - 1) / B, B, 0, stream>>>(invdeg, RR * NN);

    // x -> f16 once; weights -> transposed f16 once
    k_f32_to_f16<<<(NN * DD + B - 1) / B, B, 0, stream>>>(x, xh, NN * DD);
    {
        int t1 = RR * HH * DD;      // [R][64][128]
        k_prep_w<<<(t1 + B - 1) / B, B, 0, stream>>>(W1, wt1, DD, HH, HH, t1);
        int t2 = RR * C2PAD * HH;   // [R][48][64]
        k_prep_w<<<(t2 + B - 1) / B, B, 0, stream>>>(W2, wt2, HH, CC, C2PAD, t2);
    }

    // ---- layer 1: agg = sum_r spmm(x @ W1[r] + b1[r]) ----
    {
        const int cT = HH / 16;                 // 4
        const int tiles = mTiles * cT;          // 25000 waves
        const int blocks = (tiles * 32 + B - 1) / B;
        const int spmmTotal = EE * HH;
        for (int r = 0; r < RR; ++r) {
            k_gemm_wmma<DD><<<blocks, B, 0, stream>>>(
                xh, wt1 + (size_t)r * HH * DD, b1 + (size_t)r * HH, hbuf,
                HH, cT, tiles);
            k_spmm<HH><<<(spmmTotal + B - 1) / B, B, 0, stream>>>(
                hbuf, esrc + (size_t)r * EE, edst + (size_t)r * EE,
                invdeg + (size_t)r * NN, agg, 1.0f, spmmTotal);
        }
    }

    // h = relu(mean) -> f16 for layer-2 GEMM
    k_relu_mean_f16<<<(NN * HH + B - 1) / B, B, 0, stream>>>(agg, h16, NN * HH);

    // ---- layer 2: d_out = mean_r spmm(h @ W2[r] + b2[r]) ----
    {
        const int cT = C2PAD / 16;              // 3
        const int tiles = mTiles * cT;          // 18750 waves
        const int blocks = (tiles * 32 + B - 1) / B;
        const int spmmTotal = EE * CC;
        for (int r = 0; r < RR; ++r) {
            k_gemm_wmma<HH><<<blocks, B, 0, stream>>>(
                h16, wt2 + (size_t)r * C2PAD * HH, b2 + (size_t)r * CC, hbuf,
                CC, cT, tiles);
            k_spmm<CC><<<(spmmTotal + B - 1) / B, B, 0, stream>>>(
                hbuf, esrc + (size_t)r * EE, edst + (size_t)r * EE,
                invdeg + (size_t)r * NN, outp, 1.0f / (float)RR, spmmTotal);
        }
    }
}
